// ColorGNN_46196668236119
// MI455X (gfx1250) — matrix operand order repaired and checked by reading, not compiled, not run on hardware
//
#include <hip/hip_runtime.h>

// ColorGNN on MI455X (gfx1250, wave32):
//   - 10 dense GEMMs via v_wmma_f32_16x16x32_f16 with fused epilogues
//     (bias / bias+time-embedding / none / relu / residual), HOUT templated
//     so stores/residual loads use immediate offsets.
//   - A-tile staged to LDS with GLOBAL_LOAD_ASYNC_TO_LDS_B128 (ASYNCcnt),
//     f32->f16 conversion deferred to fragment build.
//   - GCN aggregation: degree count via float atomics, per-edge norm precompute,
//     flat (edge,channel) atomic scatter-add, self-loop term fused into agg init.

typedef __attribute__((ext_vector_type(16))) _Float16 v16h;
typedef __attribute__((ext_vector_type(8)))  float    v8f;

#define NNODES 50000
#define NEDGES 800000
#define HDIM   96
#define LA_STRIDE 100   // f32 LDS row stride for A tile (pad vs 64-bank LDS)

enum { M_BIAS = 0, M_BIAS_TE = 1, M_NONE = 2, M_RELU = 3, M_RES = 4 };

__device__ __forceinline__ unsigned lds_off32(const void* p) {
    // Generic LDS pointer = {shared_aperture, 32-bit LDS offset}; low 32 bits
    // are the raw LDS byte offset needed by async-load VDST.
    return (unsigned)(unsigned long long)p;
}

// out[row, n] = epilogue( A[row, 0:96] @ W[n, 0:96]^T )
// A: [n_rows, 96] f32, W: [HOUT, 96] f32 row-major. HOUT in {96, 32}.
template<int MODE, int HOUT>
__global__ __launch_bounds__(128)
void gemm_wmma_kernel(const float* __restrict__ A,
                      const float* __restrict__ W,
                      const float* __restrict__ bias,
                      const float* __restrict__ emb,     // M_BIAS_TE: emb_table
                      const int*   __restrict__ tptr,    // M_BIAS_TE: t index
                      const float* __restrict__ res,     // M_RES: residual [n_rows,96]
                      float* __restrict__ out,
                      int n_rows)
{
    __shared__ float    lA32[64][LA_STRIDE];  // 64-row activation tile, raw f32
    __shared__ _Float16 lW[HOUT][104];        // weight matrix, f16, padded

    const int tid  = threadIdx.x;
    const int row0 = blockIdx.x * 64;

    // ---- Async-copy A tile (64 rows x 96 f32 = 1536 x 16B chunks) into LDS.
    // One 16B async transfer per lane per iteration: no VGPR round trip,
    // consecutive lanes hit consecutive 16B chunks (fully coalesced).
    {
        const unsigned ldsbase = lds_off32(&lA32[0][0]);
#pragma unroll
        for (int it = 0; it < 12; ++it) {
            int chunk = tid + it * 128;              // 0..1535
            int r   = chunk / 24;                    // tile row
            int c16 = chunk - r * 24;                // 16B chunk within row
            int gr  = row0 + r;
            if (gr >= n_rows) gr = n_rows - 1;       // clamp tail (rows unused)
            unsigned long long gaddr =
                (unsigned long long)(const void*)(A + (size_t)gr * HDIM + c16 * 4);
            unsigned laddr = ldsbase + (unsigned)(r * LA_STRIDE + c16 * 4) * 4u;
            asm volatile("global_load_async_to_lds_b128 %0, %1, off"
                         :: "v"(laddr), "v"(gaddr) : "memory");
        }
    }

    // ---- Stage W (f32 -> f16) into LDS, float4-vectorized (rows are 96 = 24x4).
    {
        constexpr int nchunk = (HOUT * HDIM) / 4;
#pragma unroll 2
        for (int i = tid; i < nchunk; i += 128) {
            float4 w4 = ((const float4*)W)[i];
            int base = i * 4;
            int r = base / HDIM, c = base - r * HDIM;
            lW[r][c]     = (_Float16)w4.x;
            lW[r][c + 1] = (_Float16)w4.y;
            lW[r][c + 2] = (_Float16)w4.z;
            lW[r][c + 3] = (_Float16)w4.w;
        }
    }

    // Hoisted time-embedding row index (uniform scalar load).
    int trow = 0;
    if (MODE == M_BIAS_TE) trow = tptr[0];

    asm volatile("s_wait_asynccnt 0x0" ::: "memory");
    __syncthreads();

    const int wave = tid >> 5;        // wave owns rows [row0+16*wave, +16)
    const int lane = tid & 31;
    const int l16  = lane & 15;
    const int hi   = lane >> 4;       // lane half selects K sub-range

    // A fragments (16x32 f16, ISA 7.12.2 layout), converted f32->f16 here,
    // kept resident across all N tiles.
    //   k = kc*32 + (v>=4 ? 16 : 0) + 8*hi + 2*(v&3) + {0,1}
    v16h afrag[3];
#pragma unroll
    for (int kc = 0; kc < 3; ++kc) {
#pragma unroll
        for (int v = 0; v < 8; ++v) {
            int kb = kc * 32 + ((v & 4) ? 16 : 0) + hi * 8 + 2 * (v & 3);
            float2 p = *(const float2*)&lA32[wave * 16 + l16][kb];  // 8B aligned
            afrag[kc][2 * v]     = (_Float16)p.x;
            afrag[kc][2 * v + 1] = (_Float16)p.y;
        }
    }

    const bool full = (row0 + 64 <= n_rows);    // uniform: 781/782 blocks
    const int  mrow = row0 + wave * 16 + hi * 8;    // first of 8 output rows
    constexpr int NTILES = HOUT >> 4;

#pragma unroll
    for (int nt = 0; nt < NTILES; ++nt) {
        const int n = nt * 16 + l16;          // output column for this lane
        v8f acc = {0.f, 0.f, 0.f, 0.f, 0.f, 0.f, 0.f, 0.f};
#pragma unroll
        for (int kc = 0; kc < 3; ++kc) {
            // B fragment (32x16 f16): lane = column n, same K order as A;
            // VGPR v holds B[kc*32 + (v>=4 ? 16 : 0) + 8*hi + 2*(v&3) + {0,1}][n] = W[n][k]
            v16h bfrag;
#pragma unroll
            for (int v = 0; v < 8; ++v) {
                int k = kc * 32 + ((v & 4) ? 16 : 0) + hi * 8 + 2 * (v & 3);
                bfrag[2 * v]     = lW[n][k];
                bfrag[2 * v + 1] = lW[n][k + 1];
            }
            acc = __builtin_amdgcn_wmma_f32_16x16x32_f16(
                      false, afrag[kc], false, bfrag, (short)0, acc, false, false);
        }

        float extra = 0.0f;
        if (MODE != M_NONE)    extra = bias[n];
        if (MODE == M_BIAS_TE) extra += emb[(size_t)trow * HDIM + n];

        // Per-lane base pointers; v-steps become immediate instruction offsets.
        float*       po = out + (size_t)mrow * HOUT + n;
        const float* pr = (MODE == M_RES) ? (res + (size_t)mrow * HDIM + n) : nullptr;

        // C/D layout: VGPR v -> row M = v + 8*hi, column = n
        if (full) {
#pragma unroll
            for (int v = 0; v < 8; ++v) {
                float val = acc[v] + extra;
                if (MODE == M_RELU) val = fmaxf(val, 0.0f);
                if (MODE == M_RES)  val += pr[v * HDIM];
                po[v * HOUT] = val;
            }
        } else {
#pragma unroll
            for (int v = 0; v < 8; ++v) {
                if (mrow + v < n_rows) {
                    float val = acc[v] + extra;
                    if (MODE == M_RELU) val = fmaxf(val, 0.0f);
                    if (MODE == M_RES)  val += pr[v * HDIM];
                    po[v * HOUT] = val;
                }
            }
        }
    }
}

// ---------------- GCN aggregation (memory-bound) ----------------

__global__ void deg_init_kernel(float* __restrict__ deg, int n) {
    int i = blockIdx.x * blockDim.x + threadIdx.x;
    if (i < n) deg[i] = 1.0f;   // self loop
}

__global__ void deg_count_kernel(const int* __restrict__ ei, float* __restrict__ deg, int e) {
    int i = blockIdx.x * blockDim.x + threadIdx.x;
    if (i < e) atomicAdd(&deg[ei[e + i]], 1.0f);   // dst = ei[E + i]
}

// invdeg holds deg on entry; produce dis = deg^-1/2 and invdeg = 1/deg
__global__ void deg_finalize_kernel(float* __restrict__ invdeg, float* __restrict__ dis, int n) {
    int i = blockIdx.x * blockDim.x + threadIdx.x;
    if (i < n) {
        float d = invdeg[i];
        dis[i]    = rsqrtf(d);
        invdeg[i] = 1.0f / d;
    }
}

__global__ void edge_norm_kernel(const int* __restrict__ ei, const float* __restrict__ dis,
                                 float* __restrict__ norm, int e) {
    int i = blockIdx.x * blockDim.x + threadIdx.x;
    if (i < e) norm[i] = dis[ei[i]] * dis[ei[e + i]];
}

// agg = xw / deg + conv_b  (self-loop term + bias, fused)
__global__ void agg_init_kernel(const float* __restrict__ xw, const float* __restrict__ invdeg,
                                const float* __restrict__ cb, float* __restrict__ agg, int total) {
    int g = blockIdx.x * blockDim.x + threadIdx.x;
    if (g < total) {
        int i = g / HDIM;
        int c = g - i * HDIM;
        agg[g] = xw[g] * invdeg[i] + cb[c];
    }
}

// agg[dst, c] += xw[src, c] * norm[e]   (block = 4 edges x 96 channels)
__global__ void scatter_kernel(const int* __restrict__ ei, const float* __restrict__ norm,
                               const float* __restrict__ xw, float* __restrict__ agg, int e) {
    int edge = blockIdx.x * blockDim.y + threadIdx.y;
    if (edge < e) {
        int s = ei[edge];
        int d = ei[e + edge];
        float w = norm[edge];
        int c = threadIdx.x;
        atomicAdd(&agg[(size_t)d * HDIM + c], xw[(size_t)s * HDIM + c] * w);
    }
}

// ---------------- driver ----------------

extern "C" void kernel_launch(void* const* d_in, const int* in_sizes, int n_in,
                              void* d_out, int out_size, void* d_ws, size_t ws_size,
                              hipStream_t stream)
{
    (void)in_sizes; (void)n_in; (void)out_size; (void)ws_size;

    const float* x       = (const float*)d_in[0];
    const int*   t       = (const int*)  d_in[1];
    const int*   ei      = (const int*)  d_in[2];
    const float* emb     = (const float*)d_in[3];
    const float* fw_W    = (const float*)d_in[4];
    const float* fw_b    = (const float*)d_in[5];
    const float* pre_W   = (const float*)d_in[6];
    const float* pre_b   = (const float*)d_in[7];
    const float* conv_W  = (const float*)d_in[8];
    const float* conv_b  = (const float*)d_in[9];
    const float* post_W1 = (const float*)d_in[10];
    const float* post_b1 = (const float*)d_in[11];
    const float* post_W2 = (const float*)d_in[12];
    const float* post_b2 = (const float*)d_in[13];
    const float* fin_W   = (const float*)d_in[14];
    const float* fin_b   = (const float*)d_in[15];

    const int N = NNODES, E = NEDGES, H = HDIM;
    const size_t NH = (size_t)N * H;

    // Workspace layout (floats): 4*NH + 2*N + E  (~80.5 MB)
    float* ws   = (float*)d_ws;
    float* hA   = ws;              // activation ping
    float* hB   = hA + NH;         // activation pong
    float* t1   = hB + NH;         // pre_mlp out / agg buffer
    float* xw   = t1 + NH;         // conv xw / post1 out
    float* dis  = xw + NH;         // deg^-1/2        [N]
    float* invd = dis + N;         // deg (then 1/deg)[N]
    float* norm = invd + N;        // per-edge norm   [E]

    const dim3 b256(256);
    deg_init_kernel    <<<(N + 255) / 256, b256, 0, stream>>>(invd, N);
    deg_count_kernel   <<<(E + 255) / 256, b256, 0, stream>>>(ei, invd, E);
    deg_finalize_kernel<<<(N + 255) / 256, b256, 0, stream>>>(invd, dis, N);
    edge_norm_kernel   <<<(E + 255) / 256, b256, 0, stream>>>(ei, dis, norm, E);

    const int gblk = (N + 63) / 64;

    // h = x @ fw_W^T + fw_b
    gemm_wmma_kernel<M_BIAS, 96><<<gblk, 128, 0, stream>>>(
        x, fw_W, fw_b, nullptr, nullptr, nullptr, hA, N);

    float* hcur = hA;
    float* hnxt = hB;
    for (int l = 0; l < 2; ++l) {
        const size_t wo = (size_t)l * H * H;
        const size_t bo = (size_t)l * H;

        // t1 = h @ pre_W^T + pre_b + te
        gemm_wmma_kernel<M_BIAS_TE, 96><<<gblk, 128, 0, stream>>>(
            hcur, pre_W + wo, pre_b + bo, emb, t, nullptr, t1, N);

        // xw = t1 @ conv_W^T
        gemm_wmma_kernel<M_NONE, 96><<<gblk, 128, 0, stream>>>(
            t1, conv_W + wo, nullptr, nullptr, nullptr, nullptr, xw, N);

        // t1 (agg) = xw/deg + conv_b ; then scatter-add edge messages
        agg_init_kernel<<<((int)NH + 255) / 256, b256, 0, stream>>>(
            xw, invd, conv_b + bo, t1, (int)NH);
        scatter_kernel<<<(E + 3) / 4, dim3(96, 4), 0, stream>>>(ei, norm, xw, t1, E);

        // xw = relu(t1 @ post_W1^T + post_b1)
        gemm_wmma_kernel<M_RELU, 96><<<gblk, 128, 0, stream>>>(
            t1, post_W1 + wo, post_b1 + bo, nullptr, nullptr, nullptr, xw, N);

        // hnxt = xw @ post_W2^T + post_b2 + hcur (residual)
        gemm_wmma_kernel<M_RES, 96><<<gblk, 128, 0, stream>>>(
            xw, post_W2 + wo, post_b2 + bo, nullptr, nullptr, hcur, hnxt, N);

        float* tmp = hcur; hcur = hnxt; hnxt = tmp;
    }

    // out = h @ fin_W^T + fin_b   (HOUT = 32)
    gemm_wmma_kernel<M_BIAS, 32><<<gblk, 128, 0, stream>>>(
        hcur, fin_W, fin_b, nullptr, nullptr, nullptr, (float*)d_out, N);
}
